// HandwritingGenerator_28260884808174
// MI455X (gfx1250) — matrix operand order repaired and checked by reading, not compiled, not run on hardware
//
#include <hip/hip_runtime.h>

// ---------------- model constants ----------------
#define HSZ    512
#define EDIM   64
#define KMIX   10
#define BATCH  64
#define TSTEPS 700
#define ULEN   60
#define OUTC   61
#define GATES  2048
#define K1PAD  608    // [x(3) win(64) pad(29->96)] + h1(512)
#define K2PAD  1120   // [x(3) win(64) o1(512) pad(29->608)] + h2(512)
#define K3     1024
#define NWG    16
#define NTHR   256

typedef _Float16 half8   __attribute__((ext_vector_type(8)));
typedef _Float16 half16  __attribute__((ext_vector_type(16)));
typedef float    floatx8 __attribute__((ext_vector_type(8)));

// ---------------- workspace layout (bytes) ----------------
#define OFF_WC1  0ull                    // f16 [2048][608]  gate-interleaved
#define OFF_WC2  2490368ull              // f16 [2048][1120]
#define OFF_WMDN 7077888ull              // f16 [64][1024]   (rows 61..63 zero)
#define OFF_B1   7208960ull              // f32 [2048] packed bias
#define OFF_B2   7217152ull
#define OFF_H1   7225344ull              // f32 [2][64][512] double buffered
#define OFF_C1   7487488ull              // f32 [64][512]
#define OFF_H2   7618560ull              // f32 [2][64][512]
#define OFF_C2   7880704ull
#define OFF_KAP  8011776ull              // f32 [64][10]
#define OFF_WIN  8015872ull              // f32 [64][64]
#define OFF_O1   8032256ull              // f32 [64][512]
#define OFF_BAR  8163328ull              // u32 barrier counter
#define WS_NEED  8163584ull

#define SMEM_BYTES (BATCH * K2PAD * 2)   // 143360 B, largest stage (Xcat2)

__device__ __forceinline__ float sigm(float x) { return 1.0f / (1.0f + expf(-x)); }

// A fragment (16x32 f16): lane L: M = L&15, half = L>>4; K chunks at half*8 and 16+half*8
__device__ __forceinline__ half16 ld_afrag(const _Float16* base) {
    half8 lo = *(const half8*)(base);
    half8 hi = *(const half8*)(base + 16);
    return __builtin_shufflevector(lo, hi, 0,1,2,3,4,5,6,7,8,9,10,11,12,13,14,15);
}
// B fragment (32x16 f16): lane L: N = L&15, half = L>>4; K chunks at half*8 and 16+half*8
__device__ __forceinline__ half16 ld_bfrag(const _Float16* base) {
    half8 lo = *(const half8*)(base);
    half8 hi = *(const half8*)(base + 16);
    return __builtin_shufflevector(lo, hi, 0,1,2,3,4,5,6,7,8,9,10,11,12,13,14,15);
}

// C = Xlds[64 x K] * W^T(16 cols; Wp pre-offset to this wave's row block).
// Depth-2 software pipeline, 2 K-slices per rolled iteration. Loop-carried
// buffers (bf0/af0) are written exactly once per iteration with a clamped
// index, so no back-edge register copies; bf1/af1 live inside the body.
__device__ __forceinline__ void gemm_block(const _Float16* __restrict__ Wp, int wstride,
                                           const _Float16* __restrict__ Xlds, int xstride,
                                           int nkt, floatx8 acc[4], int lane) {
    const int lr = lane & 15, half = lane >> 4;
    const _Float16* wrow = Wp + (size_t)lr * wstride + half * 8;
    const _Float16* xrow = Xlds + lr * xstride + half * 8;

    half16 bf0 = ld_bfrag(wrow);
    half16 af0[4];
#pragma unroll
    for (int mt = 0; mt < 4; ++mt) af0[mt] = ld_afrag(xrow + mt * 16 * xstride);

#pragma unroll 1
    for (int kt = 0; kt < nkt - 1; kt += 2) {
        const int k1 = kt + 1;
        const int k2 = (kt + 2 < nkt) ? kt + 2 : nkt - 1;  // clamp: one dead reload at end
        // stage slice k1
        half16 bf1 = ld_bfrag(wrow + k1 * 32);
        half16 af1[4];
#pragma unroll
        for (int mt = 0; mt < 4; ++mt) af1[mt] = ld_afrag(xrow + mt * 16 * xstride + k1 * 32);
        // consume slice kt
#pragma unroll
        for (int mt = 0; mt < 4; ++mt)
            acc[mt] = __builtin_amdgcn_wmma_f32_16x16x32_f16(false, af0[mt], false, bf0,
                                                             (short)0, acc[mt], false, false);
        // stage slice k2 (overwrites consumed buffers, overlaps bf1/af1 WMMAs)
        bf0 = ld_bfrag(wrow + k2 * 32);
#pragma unroll
        for (int mt = 0; mt < 4; ++mt) af0[mt] = ld_afrag(xrow + mt * 16 * xstride + k2 * 32);
        // consume slice k1
#pragma unroll
        for (int mt = 0; mt < 4; ++mt)
            acc[mt] = __builtin_amdgcn_wmma_f32_16x16x32_f16(false, af1[mt], false, bf1,
                                                             (short)0, acc[mt], false, false);
    }
    if (nkt & 1) {      // odd trip count (19, 35): last slice already resident
#pragma unroll
        for (int mt = 0; mt < 4; ++mt)
            acc[mt] = __builtin_amdgcn_wmma_f32_16x16x32_f16(false, af0[mt], false, bf0,
                                                             (short)0, acc[mt], false, false);
    }
}

// C/D layout: VGPR r -> M = (lane>>4)*8 + r, N = lane&15
__device__ __forceinline__ void store_gates(float* LG, const floatx8 acc[4], int wave, int lane) {
    const int lr = lane & 15, half = lane >> 4;
#pragma unroll
    for (int mt = 0; mt < 4; ++mt)
#pragma unroll
        for (int r = 0; r < 8; ++r)
            LG[(mt * 16 + half * 8 + r) * 128 + wave * 16 + lr] = acc[mt][r];
}

__device__ __forceinline__ void lstm_update(const float* LG, const float* biasp,
                                            float* cw, float* hnew, int wg, int tid) {
    for (int e = tid; e < 2048; e += NTHR) {
        const int b = e >> 5, u = e & 31;
        const float* lg = LG + b * 128 + 4 * u;
        const float* bs = biasp + wg * 128 + 4 * u;
        float gi = lg[0] + bs[0], gf = lg[1] + bs[1];
        float gg = lg[2] + bs[2], go = lg[3] + bs[3];
        const int h = b * HSZ + wg * 32 + u;
        float c = cw[h];
        float cn = sigm(gf) * c + sigm(gi) * tanhf(gg);
        cw[h] = cn;
        hnew[h] = sigm(go) * tanhf(cn);
    }
}

__device__ __forceinline__ void grid_barrier(unsigned* bar, unsigned target) {
    __threadfence();
    __syncthreads();
    if (threadIdx.x == 0) {
        __hip_atomic_fetch_add(bar, 1u, __ATOMIC_RELEASE, __HIP_MEMORY_SCOPE_AGENT);
        while (__hip_atomic_load(bar, __ATOMIC_ACQUIRE, __HIP_MEMORY_SCOPE_AGENT) < target)
            __builtin_amdgcn_s_sleep(2);
    }
    __syncthreads();
    __threadfence();
}

// ---------------- prep: zero state + barrier ----------------
__global__ void hw_init_state(unsigned char* ws) {
    float* p = (float*)(ws + OFF_H1);
    const size_t n = (WS_NEED - OFF_H1) / 4;   // h1,c1,h2,c2,kappa,window,o1,barrier
    const size_t stride = (size_t)gridDim.x * blockDim.x;
    for (size_t i = (size_t)blockIdx.x * blockDim.x + threadIdx.x; i < n; i += stride)
        p[i] = 0.0f;
}

// ---------------- prep: pack weights to f16, gate-interleaved ----------------
__global__ void hw_pack_weights(const float* __restrict__ W_ih1, const float* __restrict__ W_hh1,
                                const float* __restrict__ b_ih1, const float* __restrict__ b_hh1,
                                const float* __restrict__ W_ih2, const float* __restrict__ W_hh2,
                                const float* __restrict__ b_ih2, const float* __restrict__ b_hh2,
                                const float* __restrict__ W_mdn, unsigned char* ws) {
    _Float16* Wc1 = (_Float16*)(ws + OFF_WC1);
    _Float16* Wc2 = (_Float16*)(ws + OFF_WC2);
    _Float16* Wm  = (_Float16*)(ws + OFF_WMDN);
    float* bias1  = (float*)(ws + OFF_B1);
    float* bias2  = (float*)(ws + OFF_B2);
    const long stride = (long)gridDim.x * blockDim.x;
    const long gid = (long)blockIdx.x * blockDim.x + threadIdx.x;

    for (long i = gid; i < (long)GATES * K1PAD; i += stride) {
        int r = (int)(i / K1PAD), k = (int)(i % K1PAD);
        int orig = (r & 3) * HSZ + (r >> 2);
        float v = (k < 67) ? W_ih1[orig * 67 + k]
                : (k < 96) ? 0.0f
                           : W_hh1[orig * HSZ + (k - 96)];
        Wc1[i] = (_Float16)v;
    }
    for (long i = gid; i < (long)GATES * K2PAD; i += stride) {
        int r = (int)(i / K2PAD), k = (int)(i % K2PAD);
        int orig = (r & 3) * HSZ + (r >> 2);
        float v = (k < 579) ? W_ih2[orig * 579 + k]
                : (k < 608) ? 0.0f
                            : W_hh2[orig * HSZ + (k - 608)];
        Wc2[i] = (_Float16)v;
    }
    for (long i = gid; i < 64L * K3; i += stride) {
        int n = (int)(i >> 10), k = (int)(i & 1023);
        Wm[i] = (_Float16)((n < OUTC) ? W_mdn[n * K3 + k] : 0.0f);
    }
    for (long r = gid; r < GATES; r += stride) {
        int orig = ((int)r & 3) * HSZ + ((int)r >> 2);
        bias1[r] = b_ih1[orig] + b_hh1[orig];
        bias2[r] = b_ih2[orig] + b_hh2[orig];
    }
}

// ---------------- persistent step kernel ----------------
__global__ __launch_bounds__(NTHR, 1)
void hw_persistent(const float* __restrict__ strokes, const int* __restrict__ char_idx,
                   const float* __restrict__ embed, const float* __restrict__ W_proj,
                   const float* __restrict__ b_proj,
                   const float* __restrict__ g1, const float* __restrict__ be1,
                   const float* __restrict__ g2, const float* __restrict__ be2,
                   const float* __restrict__ b_mdn,
                   unsigned char* __restrict__ ws, float* __restrict__ out) {
    const int wg = blockIdx.x, tid = threadIdx.x;
    const int wave = tid >> 5, lane = tid & 31;
    const int lr = lane & 15, half = lane >> 4;

    __shared__ __align__(16) unsigned char SMEM[SMEM_BYTES];
    _Float16* X = (_Float16*)SMEM;

    const _Float16* Wc1 = (const _Float16*)(ws + OFF_WC1);
    const _Float16* Wc2 = (const _Float16*)(ws + OFF_WC2);
    const _Float16* Wm  = (const _Float16*)(ws + OFF_WMDN);
    const float* bias1  = (const float*)(ws + OFF_B1);
    const float* bias2  = (const float*)(ws + OFF_B2);
    float* h1buf = (float*)(ws + OFF_H1);
    float* c1w   = (float*)(ws + OFF_C1);
    float* h2buf = (float*)(ws + OFF_H2);
    float* c2w   = (float*)(ws + OFF_C2);
    float* kapw  = (float*)(ws + OFF_KAP);
    float* winw  = (float*)(ws + OFF_WIN);
    float* o1w   = (float*)(ws + OFF_O1);
    unsigned* bar = (unsigned*)(ws + OFF_BAR);

    unsigned ep = 0;

    for (int t = 0; t < TSTEPS; ++t) {
        const float* h1p = h1buf + (t & 1) * (BATCH * HSZ);
        float*       h1c = h1buf + ((t + 1) & 1) * (BATCH * HSZ);
        const float* h2p = h2buf + (t & 1) * (BATCH * HSZ);
        float*       h2c = h2buf + ((t + 1) & 1) * (BATCH * HSZ);

        // ======== Xcat1 = [x_t | window | 0 | h1_prev] -> LDS f16 ========
        {
            const int b = tid >> 2, q = tid & 3;
            const float* xs = strokes + (b * TSTEPS + t) * 3;
            for (int k = q; k < K1PAD; k += 4) {
                float v = (k < 3)  ? xs[k]
                        : (k < 67) ? winw[b * EDIM + (k - 3)]
                        : (k < 96) ? 0.0f
                                   : h1p[b * HSZ + (k - 96)];
                X[b * K1PAD + k] = (_Float16)v;
            }
        }
        __syncthreads();

        // ======== GEMM1: gates1[64,2048], WG owns 128 cols, wave owns 16 ========
        {
            floatx8 z = {};
            floatx8 acc[4] = {z, z, z, z};
            gemm_block(Wc1 + (size_t)(wg * 128 + wave * 16) * K1PAD, K1PAD,
                       X, K1PAD, K1PAD / 32, acc, lane);
            __syncthreads();                       // Xcat1 dead; alias LDS as gates
            store_gates((float*)SMEM, acc, wave, lane);
        }
        __syncthreads();
        lstm_update((const float*)SMEM, bias1, c1w, h1c, wg, tid);
        grid_barrier(bar, NWG * (++ep));           // h1(t) visible everywhere

        // ======== attention / layernorm: 4 batch rows per WG ========
        {
            float* o1sm = (float*)SMEM;            // [4][512]
            float* red  = o1sm + 4 * 512;          // [2][4][64]
            float* raws = red + 2 * 4 * 64;        // [4][32]
            float* alps = raws + 4 * 32;           // [4][16] x3
            float* bets = alps + 4 * 16;
            float* kaps = bets + 4 * 16;
            float* phis = kaps + 4 * 16;           // [4][64]
            float* musg = phis + 4 * 64;           // [4][2]
            const int bl = tid >> 6, l = tid & 63;
            const int b = wg * 4 + bl;

            float s1 = 0.f, s2 = 0.f;
            for (int j = l; j < HSZ; j += 64) { float v = h1c[b * HSZ + j]; s1 += v; s2 += v * v; }
            red[bl * 64 + l] = s1; red[256 + bl * 64 + l] = s2;
            __syncthreads();
            if (l == 0) {
                float a = 0.f, c = 0.f;
                for (int i = 0; i < 64; ++i) { a += red[bl * 64 + i]; c += red[256 + bl * 64 + i]; }
                float mu = a / HSZ, var = c / HSZ - mu * mu;
                musg[bl * 2] = mu; musg[bl * 2 + 1] = rsqrtf(var + 1e-5f);
            }
            __syncthreads();
            const float mu = musg[bl * 2], rs = musg[bl * 2 + 1];
            for (int j = l; j < HSZ; j += 64) {
                float v = (h1c[b * HSZ + j] - mu) * rs * g1[j] + be1[j];
                o1sm[bl * 512 + j] = v;
                o1w[b * HSZ + j] = v;
            }
            __syncthreads();
            if (l < 3 * KMIX) {                    // projection: 30 dots of 512
                float d = b_proj[l];
                const float* wp = W_proj + l * HSZ;
                for (int j = 0; j < HSZ; ++j) d += o1sm[bl * 512 + j] * wp[j];
                raws[bl * 32 + l] = d;
            }
            __syncthreads();
            if (l < KMIX) {
                float al = expf(raws[bl * 32 + l]);
                float be = expf(raws[bl * 32 + 10 + l]);
                float ka = kapw[b * KMIX + l] + expf(raws[bl * 32 + 20 + l]);
                kapw[b * KMIX + l] = ka;
                alps[bl * 16 + l] = al; bets[bl * 16 + l] = be; kaps[bl * 16 + l] = ka;
            }
            __syncthreads();
            if (l < ULEN) {
                float s = 0.f;
                for (int k = 0; k < KMIX; ++k) {
                    float d = kaps[bl * 16 + k] - (float)l;
                    s += alps[bl * 16 + k] * expf(-bets[bl * 16 + k] * d * d);
                }
                phis[bl * 64 + l] = s;
            }
            __syncthreads();
            {                                       // window[b, e=l]
                float s = 0.f;
                const int* ci = char_idx + b * ULEN;
                for (int u = 0; u < ULEN; ++u) s += phis[bl * 64 + u] * embed[ci[u] * EDIM + l];
                winw[b * EDIM + l] = s;
            }
        }
        grid_barrier(bar, NWG * (++ep));           // window(t), o1(t) visible

        // ======== Xcat2 = [x_t | window | o1 | 0 | h2_prev] ========
        {
            const int b = tid >> 2, q = tid & 3;
            const float* xs = strokes + (b * TSTEPS + t) * 3;
            for (int k = q; k < K2PAD; k += 4) {
                float v = (k < 3)   ? xs[k]
                        : (k < 67)  ? winw[b * EDIM + (k - 3)]
                        : (k < 579) ? o1w[b * HSZ + (k - 67)]
                        : (k < 608) ? 0.0f
                                    : h2p[b * HSZ + (k - 608)];
                X[b * K2PAD + k] = (_Float16)v;
            }
        }
        __syncthreads();

        // ======== GEMM2 + LSTM2 update ========
        {
            floatx8 z = {};
            floatx8 acc[4] = {z, z, z, z};
            gemm_block(Wc2 + (size_t)(wg * 128 + wave * 16) * K2PAD, K2PAD,
                       X, K2PAD, K2PAD / 32, acc, lane);
            __syncthreads();
            store_gates((float*)SMEM, acc, wave, lane);
        }
        __syncthreads();
        lstm_update((const float*)SMEM, bias2, c2w, h2c, wg, tid);
        grid_barrier(bar, NWG * (++ep));           // h2(t) visible

        // ======== MDN head: C-tile (mt,nt) per WG; o2 = LN(h2) built locally ========
        {
            const int mt = wg >> 2, nt = wg & 3;
            _Float16* A3 = (_Float16*)SMEM;                    // [16][1024]
            float* partC = (float*)(SMEM + 32768);             // [8][256]
            float* red3  = (float*)(SMEM + 32768 + 8192);      // [16][16][2]
            float* mus3  = red3 + 16 * 16 * 2;                 // [16][2]
            const int r = tid >> 4, s = tid & 15;
            const int b = mt * 16 + r;

            float s1 = 0.f, s2 = 0.f;
            for (int j = s * 32; j < s * 32 + 32; ++j) { float v = h2c[b * HSZ + j]; s1 += v; s2 += v * v; }
            red3[(r * 16 + s) * 2] = s1; red3[(r * 16 + s) * 2 + 1] = s2;
            __syncthreads();
            if (s == 0) {
                float a = 0.f, c = 0.f;
                for (int i = 0; i < 16; ++i) { a += red3[(r * 16 + i) * 2]; c += red3[(r * 16 + i) * 2 + 1]; }
                float mu = a / HSZ, var = c / HSZ - mu * mu;
                mus3[r * 2] = mu; mus3[r * 2 + 1] = rsqrtf(var + 1e-5f);
            }
            __syncthreads();
            const float mu = mus3[r * 2], rs = mus3[r * 2 + 1];
            for (int c = s * 64; c < s * 64 + 64; ++c) {
                float v;
                if (c < HSZ) v = o1w[b * HSZ + c];
                else { int j = c - HSZ; v = (h2c[b * HSZ + j] - mu) * rs * g2[j] + be2[j]; }
                A3[r * 1024 + c] = (_Float16)v;
            }
            __syncthreads();

            // all 4 K-slices fetched up-front (one clause), then 4 chained WMMAs
            const _Float16* wrow = Wm + (size_t)(nt * 16 + lr) * K3 + half * 8;
            half16 bfv[4], afv[4];
#pragma unroll
            for (int i = 0; i < 4; ++i) {
                const int kt = wave + i * 8;
                bfv[i] = ld_bfrag(wrow + kt * 32);
                afv[i] = ld_afrag(A3 + lr * 1024 + kt * 32 + half * 8);
            }
            floatx8 a3 = {};
#pragma unroll
            for (int i = 0; i < 4; ++i)
                a3 = __builtin_amdgcn_wmma_f32_16x16x32_f16(false, afv[i], false, bfv[i],
                                                            (short)0, a3, false, false);
#pragma unroll
            for (int rr = 0; rr < 8; ++rr)
                partC[wave * 256 + (half * 8 + rr) * 16 + lr] = a3[rr];
            __syncthreads();
            {
                const int m = tid >> 4, n = tid & 15;
                float sum = 0.f;
                for (int w2 = 0; w2 < 8; ++w2) sum += partC[w2 * 256 + m * 16 + n];
                const int col = nt * 16 + n;
                if (col < OUTC)
                    out[((mt * 16 + m) * TSTEPS + t) * OUTC + col] = sum + b_mdn[col];
            }
            __syncthreads();                                    // SMEM reused next step
        }
    }
}

extern "C" void kernel_launch(void* const* d_in, const int* in_sizes, int n_in,
                              void* d_out, int out_size, void* d_ws, size_t ws_size,
                              hipStream_t stream) {
    const float* strokes = (const float*)d_in[0];
    const int*   char_idx = (const int*)d_in[1];
    const float* embed   = (const float*)d_in[2];
    const float* W_proj  = (const float*)d_in[3];
    const float* b_proj  = (const float*)d_in[4];
    const float* W_ih1   = (const float*)d_in[5];
    const float* W_hh1   = (const float*)d_in[6];
    const float* b_ih1   = (const float*)d_in[7];
    const float* b_hh1   = (const float*)d_in[8];
    const float* W_ih2   = (const float*)d_in[9];
    const float* W_hh2   = (const float*)d_in[10];
    const float* b_ih2   = (const float*)d_in[11];
    const float* b_hh2   = (const float*)d_in[12];
    const float* g1      = (const float*)d_in[13];
    const float* be1     = (const float*)d_in[14];
    const float* g2      = (const float*)d_in[15];
    const float* be2     = (const float*)d_in[16];
    const float* W_mdn   = (const float*)d_in[17];
    const float* b_mdn   = (const float*)d_in[18];
    unsigned char* ws = (unsigned char*)d_ws;
    float* out = (float*)d_out;

    hipLaunchKernelGGL(hw_init_state, dim3(64), dim3(NTHR), 0, stream, ws);
    hipLaunchKernelGGL(hw_pack_weights, dim3(512), dim3(NTHR), 0, stream,
                       W_ih1, W_hh1, b_ih1, b_hh1, W_ih2, W_hh2, b_ih2, b_hh2, W_mdn, ws);
    hipLaunchKernelGGL(hw_persistent, dim3(NWG), dim3(NTHR), 0, stream,
                       strokes, char_idx, embed, W_proj, b_proj,
                       g1, be1, g2, be2, b_mdn, ws, out);
}
